// GatedAttentionalAssociationUnit_81243601371202
// MI455X (gfx1250) — matrix operand order repaired and checked by reading, not compiled, not run on hardware
//
#include <hip/hip_runtime.h>

// ---------------------------------------------------------------------------
// GatedAttentionalAssociationUnit for MI455X (gfx1250, wave32, WMMA).
// B=8, N=2048, IN=128, HID=64.
//
// Roofline: adj stream (134MB) dominates bytes (~5.8us @ 23.3TB/s); the
// attention@Wh GEMM (4.3 GFLOP) runs on v_wmma_f32_16x16x32_f16 to hide
// under it. Masked softmax uses the analytic bound
// c_i = lrelu(Wh1_i + max_j Wh2_j) (lrelu monotone) so adj is read exactly
// once with no online rescaling. The B operand (Wh tile) is staged once per
// workgroup into LDS via async global->LDS DMA (ASYNCcnt), double-buffered
// one j-block ahead, instead of 8 redundant per-wave global fetches.
// ---------------------------------------------------------------------------

#define B_      8
#define N_      2048
#define IN_DIM_ 128
#define HID_    64
#define KTOT    192   // IN_DIM + HID (fused gating K)
#define GATE3   192   // 3*HID

typedef _Float16 v16h __attribute__((ext_vector_type(16)));
typedef _Float16 v8h  __attribute__((ext_vector_type(8)));
typedef float    v8f  __attribute__((ext_vector_type(8)));

union Frag16 { v16h v; v8h h[2]; };

static __device__ __forceinline__ v8f wmma_f16(v16h a, v16h b, v8f c) {
  // D = A(16x32 f16) * B(32x16 f16) + C(16x16 f32)
  return __builtin_amdgcn_wmma_f32_16x16x32_f16(false, a, false, b,
                                                (short)0, c, false, false);
}

// Async DMA: copy 16B from global to LDS, tracked by ASYNCcnt.
// vdst = LDS byte offset (low 32 bits of the generic shared pointer).
static __device__ __forceinline__ void async_load_b128(void* lds, const void* gptr) {
  unsigned int loff = (unsigned int)(unsigned long long)lds;
  unsigned long long ga = (unsigned long long)gptr;
  asm volatile("global_load_async_to_lds_b128 %0, %1, off"
               :: "v"(loff), "v"(ga) : "memory");
}

static __device__ __forceinline__ void wait_async0() {
  asm volatile("s_wait_asynccnt 0" ::: "memory");
}

// --------------------------- pack kernels ----------------------------------

__global__ void k0_pack_weights(const float* __restrict__ X2H,
                                const float* __restrict__ H2H,
                                _Float16* __restrict__ Wc16) {
  int i = blockIdx.x * blockDim.x + threadIdx.x;   // over 192*192
  if (i >= KTOT * GATE3) return;
  int k = i / GATE3, c = i % GATE3;
  float v = (k < IN_DIM_) ? X2H[k * GATE3 + c] : H2H[(k - IN_DIM_) * GATE3 + c];
  Wc16[c * KTOT + k] = (_Float16)v;
}

__global__ void k0_pack_inputs(const float* __restrict__ x,
                               const float* __restrict__ h,
                               _Float16* __restrict__ Xc16) {
  long long i = (long long)blockIdx.x * blockDim.x + threadIdx.x;  // B*N*192
  if (i >= (long long)B_ * N_ * KTOT) return;
  int c = (int)(i % KTOT);
  long long row = i / KTOT;
  float v = (c < IN_DIM_) ? x[row * IN_DIM_ + c] : h[row * HID_ + (c - IN_DIM_)];
  Xc16[i] = (_Float16)v;
}

// --------------------------- gated update ----------------------------------
// One wave = one 16-row tile. 4 waves / block, 256 blocks -> 1024 tiles.
// A fragment (16x32 f16): lane L holds row M=L%16; halves 0..7 -> K=8g..8g+7,
// halves 8..15 -> K=16+8g.. (g = L/16).  (ISA 7.12.2)
// B fragment (32x16 f16): lane L holds N=L%16; halves same K order as A.

__global__ __launch_bounds__(128) void k1_gate(
    const _Float16* __restrict__ Xc16, const _Float16* __restrict__ Wc16,
    const float* __restrict__ hbuf, const float* __restrict__ avec,
    float* __restrict__ Wh_out, _Float16* __restrict__ Wh16,
    float* __restrict__ Wh1, float* __restrict__ Wh2) {
  const int lane = threadIdx.x & 31;
  const int wv   = threadIdx.x >> 5;
  const int tile = blockIdx.x * 4 + wv;           // 0..1023 (row-major over B*N)
  const int g    = lane >> 4;
  const int lr   = lane & 15;
  const long long row0 = (long long)tile * 16;

  // Load the 6 A fragments (K = 192 = 6*32) once; reused by all 12 N-tiles.
  Frag16 A[6];
  const long long abase = (row0 + lr) * KTOT;
  #pragma unroll
  for (int ks = 0; ks < 6; ++ks) {
    const _Float16* p = Xc16 + abase + ks * 32 + 8 * g;
    A[ks].h[0] = *(const v8h*)p;
    A[ks].h[1] = *(const v8h*)(p + 16);
  }

  v8f accRG[4], accIG[4], accNI[4], accNH[4];

  // cols 0..127: resetgate/inputgate pre-activations = x@X2H + h@H2H (sum ok)
  #pragma unroll
  for (int nt = 0; nt < 8; ++nt) {
    v8f acc = {};
    #pragma unroll
    for (int ks = 0; ks < 6; ++ks) {
      Frag16 Bf;
      const _Float16* p = Wc16 + (nt * 16 + lr) * KTOT + ks * 32 + 8 * g;
      Bf.h[0] = *(const v8h*)p;
      Bf.h[1] = *(const v8h*)(p + 16);
      acc = wmma_f16(A[ks].v, Bf.v, acc);
    }
    if (nt < 4) accRG[nt] = acc; else accIG[nt - 4] = acc;
  }
  // cols 128..191: i_n (K over x part only) and h_n (K over h part only)
  #pragma unroll
  for (int nt = 0; nt < 4; ++nt) {
    v8f ai = {}, ah = {};
    #pragma unroll
    for (int ks = 0; ks < 4; ++ks) {
      Frag16 Bf;
      const _Float16* p = Wc16 + ((8 + nt) * 16 + lr) * KTOT + ks * 32 + 8 * g;
      Bf.h[0] = *(const v8h*)p;
      Bf.h[1] = *(const v8h*)(p + 16);
      ai = wmma_f16(A[ks].v, Bf.v, ai);
    }
    #pragma unroll
    for (int ks = 4; ks < 6; ++ks) {
      Frag16 Bf;
      const _Float16* p = Wc16 + ((8 + nt) * 16 + lr) * KTOT + ks * 32 + 8 * g;
      Bf.h[0] = *(const v8h*)p;
      Bf.h[1] = *(const v8h*)(p + 16);
      ah = wmma_f16(A[ks].v, Bf.v, ah);
    }
    accNI[nt] = ai;
    accNH[nt] = ah;
  }

  // C/D layout: lane L, vgpr r -> row = r + 8*g, col = t*16 + lr
  v8f irh[4], ig4[4];
  #pragma unroll
  for (int t = 0; t < 4; ++t) {
    #pragma unroll
    for (int r = 0; r < 8; ++r) {
      float rg = 1.0f / (1.0f + __expf(-accRG[t][r]));
      ig4[t][r] = 1.0f / (1.0f + __expf(-accIG[t][r]));
      irh[t][r] = accNI[t][r] + rg * accNH[t][r];
    }
  }

  // LayerNorm over the 64 hidden cols per row: reduce across 16 lanes (xor<=8)
  float mu[8], rstd[8];
  #pragma unroll
  for (int r = 0; r < 8; ++r) {
    float s  = irh[0][r] + irh[1][r] + irh[2][r] + irh[3][r];
    float s2 = irh[0][r] * irh[0][r] + irh[1][r] * irh[1][r] +
               irh[2][r] * irh[2][r] + irh[3][r] * irh[3][r];
    #pragma unroll
    for (int m = 8; m >= 1; m >>= 1) {
      s  += __shfl_xor(s, m, 32);
      s2 += __shfl_xor(s2, m, 32);
    }
    mu[r] = s * (1.0f / 64.0f);
    float var = s2 * (1.0f / 64.0f) - mu[r] * mu[r];
    rstd[r] = rsqrtf(var + 1e-5f);
  }

  float a1[4], a2[4];
  #pragma unroll
  for (int t = 0; t < 4; ++t) {
    a1[t] = avec[t * 16 + lr];
    a2[t] = avec[64 + t * 16 + lr];
  }

  float s1[8], s2v[8];
  #pragma unroll
  for (int r = 0; r < 8; ++r) { s1[r] = 0.f; s2v[r] = 0.f; }

  #pragma unroll
  for (int t = 0; t < 4; ++t) {
    #pragma unroll
    for (int r = 0; r < 8; ++r) {
      float ng = tanhf((irh[t][r] - mu[r]) * rstd[r]);
      long long grow = row0 + r + 8 * g;
      float hv = hbuf[grow * HID_ + t * 16 + lr];
      float wh = ng + ig4[t][r] * (hv - ng);
      Wh_out[grow * HID_ + t * 16 + lr] = wh;       // 2nd output (f32)
      Wh16[grow * HID_ + t * 16 + lr] = (_Float16)wh;
      s1[r] += wh * a1[t];
      s2v[r] += wh * a2[t];
    }
  }
  #pragma unroll
  for (int r = 0; r < 8; ++r) {
    #pragma unroll
    for (int m = 8; m >= 1; m >>= 1) {
      s1[r]  += __shfl_xor(s1[r], m, 32);
      s2v[r] += __shfl_xor(s2v[r], m, 32);
    }
    if (lr == 0) {
      long long grow = row0 + r + 8 * g;
      Wh1[grow] = s1[r];
      Wh2[grow] = s2v[r];
    }
  }
}

// per-batch max_j Wh2[j] (for the overflow-safe softmax bound)
__global__ void k1b_maxw2(const float* __restrict__ Wh2, float* __restrict__ Wmax2) {
  __shared__ float red[256];
  int b = blockIdx.x, t = threadIdx.x;
  float m = -3.4e38f;
  for (int j = t; j < N_; j += 256) m = fmaxf(m, Wh2[b * N_ + j]);
  red[t] = m;
  __syncthreads();
  for (int s = 128; s > 0; s >>= 1) {
    if (t < s) red[t] = fmaxf(red[t], red[t + s]);
    __syncthreads();
  }
  if (t == 0) Wmax2[b] = red[0];
}

// --------------------------- masked attention ------------------------------
// Block = 8 waves; wave handles 16 i-rows, streams adj once, builds f16 P
// tiles in LDS, accumulates P@Wh with 4 WMMAs per 32-wide j-block.
// The shared 32x64 f16 Wh tile is DMA'd into LDS (async, double-buffered).

#define PSTR 72  // halves per P-row (144B: 16B aligned, bank-friendly)
#define BSTR 72  // halves per staged-Wh row (144B)

__global__ __launch_bounds__(256) void k2_attn(
    const int* __restrict__ adj, const float* __restrict__ Wh1,
    const float* __restrict__ Wh2, const float* __restrict__ Wmax2,
    const _Float16* __restrict__ Wh16, float* __restrict__ hprime) {
  __shared__ float    wh2s[N_];
  __shared__ _Float16 p16[8][16 * PSTR];
  __shared__ float    linv[8][16];
  __shared__ _Float16 whs[2][32 * BSTR];   // double-buffered B tile

  const int tid = threadIdx.x, lane = tid & 31, wv = tid >> 5;
  const int b = blockIdx.x >> 4, iblk = blockIdx.x & 15;
  const int i0 = iblk * 128 + wv * 16;
  const int g = lane >> 4, lr = lane & 15;

  for (int j = tid * 4; j < N_; j += 256 * 4)
    *(float4*)&wh2s[j] = *(const float4*)&Wh2[b * N_ + j];

  const float wmax = Wmax2[b];
  const float wh1v = Wh1[(long long)b * N_ + i0 + lr];
  float sc = wh1v + wmax;
  const float cval = sc > 0.f ? sc : 0.2f * sc;   // lrelu(upper bound)
  float lacc = 0.f;
  v8f acc[4] = {};

  // async staging slot: 256 threads cover 32 rows x 8 chunks of 8 halves
  const int srow = tid >> 3, schk = tid & 7;
  const _Float16* wh16b = Wh16 + ((long long)b * N_ + srow) * HID_ + schk * 8;
  // prefill buffer 0 with the j0=0 tile
  async_load_b128(&whs[0][srow * BSTR + schk * 8], wh16b);

  __syncthreads();   // wh2s ready

  const int* adjbase = adj + ((long long)b * N_ + i0 + lr) * N_;
  _Float16* prow = &p16[wv][lr * PSTR];

  for (int j0 = 0; j0 < N_; j0 += 32) {
    const int cur = (j0 >> 5) & 1;
    const int jb = j0 + g * 16;
    const int4* arow = (const int4*)(adjbase + jb);
    int4 m0 = arow[0], m1 = arow[1], m2 = arow[2], m3 = arow[3];
    const float4* wrow = (const float4*)&wh2s[jb];
    float4 w0 = wrow[0], w1 = wrow[1], w2 = wrow[2], w3 = wrow[3];

    Frag16 P;
#define DOP(mask, w, slot)                                            \
    {                                                                 \
      float s_ = wh1v + (w);                                          \
      float e_ = s_ > 0.f ? s_ : 0.2f * s_;                           \
      float p_ = (mask) ? __expf(e_ - cval) : 0.f;                    \
      lacc += p_;                                                     \
      P.v[slot] = (_Float16)p_;                                       \
    }
    DOP(m0.x, w0.x, 0)  DOP(m0.y, w0.y, 1)  DOP(m0.z, w0.z, 2)  DOP(m0.w, w0.w, 3)
    DOP(m1.x, w1.x, 4)  DOP(m1.y, w1.y, 5)  DOP(m1.z, w1.z, 6)  DOP(m1.w, w1.w, 7)
    DOP(m2.x, w2.x, 8)  DOP(m2.y, w2.y, 9)  DOP(m2.z, w2.z, 10) DOP(m2.w, w2.w, 11)
    DOP(m3.x, w3.x, 12) DOP(m3.y, w3.y, 13) DOP(m3.z, w3.z, 14) DOP(m3.w, w3.w, 15)
#undef DOP

    // stage P row-major in LDS (lane writes row lr, cols g*16..g*16+15)
    *(v8h*)(prow + g * 16)     = P.h[0];
    *(v8h*)(prow + g * 16 + 8) = P.h[1];
    asm volatile("s_wait_dscnt 0" ::: "memory");  // cross-lane LDS handoff

    // A fragment from staged P tile
    Frag16 Af;
    const _Float16* ap = &p16[wv][lr * PSTR + 8 * g];
    Af.h[0] = *(const v8h*)ap;
    Af.h[1] = *(const v8h*)(ap + 16);

    // make the DMA'd B tile visible to all waves, then kick off the next one
    wait_async0();
    __syncthreads();
    if (j0 + 32 < N_)
      async_load_b128(&whs[cur ^ 1][srow * BSTR + schk * 8],
                      wh16b + (long long)(j0 + 32) * HID_);

    // B fragments from LDS; lane L -> col lr, K-rows 8g..8g+7, 16+8g..16+8g+7
    const _Float16* bbase = &whs[cur][8 * g * BSTR + lr];
    #pragma unroll
    for (int nt = 0; nt < 4; ++nt) {
      Frag16 Bf;
      #pragma unroll
      for (int s = 0; s < 8; ++s) {
        Bf.v[s]     = bbase[s * BSTR + nt * 16];
        Bf.v[8 + s] = bbase[(16 + s) * BSTR + nt * 16];
      }
      acc[nt] = wmma_f16(Af.v, Bf.v, acc[nt]);
    }

    if (j0 + 32 < N_)
      __builtin_prefetch(adjbase + j0 + 32 + g * 16, 0, 1);  // global_prefetch_b8
  }

  // combine the two half-wave partial row sums, normalize, store
  lacc += __shfl_xor(lacc, 16, 32);
  if (g == 0) linv[wv][lr] = 1.0f / lacc;
  asm volatile("s_wait_dscnt 0" ::: "memory");

  #pragma unroll
  for (int nt = 0; nt < 4; ++nt) {
    #pragma unroll
    for (int r = 0; r < 8; ++r) {
      long long orow = (long long)b * N_ + i0 + r + 8 * g;
      hprime[orow * HID_ + nt * 16 + lr] = acc[nt][r] * linv[wv][r + 8 * g];
    }
  }
}

// --------------------------- launcher --------------------------------------

extern "C" void kernel_launch(void* const* d_in, const int* in_sizes, int n_in,
                              void* d_out, int out_size, void* d_ws, size_t ws_size,
                              hipStream_t stream) {
  const int*   adj = (const int*)d_in[0];
  const float* x   = (const float*)d_in[1];
  const float* h   = (const float*)d_in[2];
  const float* X2H = (const float*)d_in[3];
  const float* H2H = (const float*)d_in[4];
  const float* a   = (const float*)d_in[5];
  float* out = (float*)d_out;
  char*  ws  = (char*)d_ws;

  // workspace layout (bytes, 256-aligned)
  _Float16* Wc16 = (_Float16*)(ws + 0);         //   73,728
  _Float16* Xc16 = (_Float16*)(ws + 73728);     // 6,291,456
  _Float16* Wh16 = (_Float16*)(ws + 6365184);   // 2,097,152
  float*    Wh1  = (float*)(ws + 8462336);      //    65,536
  float*    Wh2  = (float*)(ws + 8527872);      //    65,536
  float*    Wmax = (float*)(ws + 8593408);      //        32

  float* hprime = out;                               // output 0
  float* Wh_out = out + (size_t)B_ * N_ * HID_;      // output 1

  k0_pack_weights<<<(KTOT * GATE3 + 255) / 256, 256, 0, stream>>>(X2H, H2H, Wc16);
  long long nX = (long long)B_ * N_ * KTOT;
  k0_pack_inputs<<<(int)((nX + 255) / 256), 256, 0, stream>>>(x, h, Xc16);
  k1_gate<<<256, 128, 0, stream>>>(Xc16, Wc16, h, a, Wh_out, Wh16, Wh1, Wh2);
  k1b_maxw2<<<B_, 256, 0, stream>>>(Wh2, Wmax);
  k2_attn<<<B_ * (N_ / 128), 256, 0, stream>>>(adj, Wh1, Wh2, Wmax, Wh16, hprime);
}
